// SAPADeform_78563541778854
// MI455X (gfx1250) — matrix operand order repaired and checked by reading, not compiled, not run on hardware
//
#include <hip/hip_runtime.h>
#include <hip/hip_bf16.h>
#include <math.h>

typedef __bf16 bf16_t;
typedef bf16_t v16bf __attribute__((ext_vector_type(16)));
typedef float  v8f   __attribute__((ext_vector_type(8)));

#define NGROUPS 4
#define NPOINT  9
#define EDIM    32
#define CIN     256
#define H_LO    64
#define W_LO    64
#define HW_LO   (H_LO * W_LO)
#define H_UP    128
#define W_UP    128
#define HW_UP   (H_UP * W_UP)
#define OFFCH   288   // NGROUPS * NPOINT * 2 * UP * UP

// -----------------------------------------------------------------------------
// 1x1 conv as GEMM via WMMA bf16: out[b,m,p] = sum_c Wm[m,c] * img[b,c,p] + bias[m]
// One wave computes a 32(M) x 16(N-pixels) fp32 supertile (two 16x16 WMMA tiles
// sharing one B fragment), K-loop over CIN in steps of 32.
// HW is a compile-time constant so all channel-strided accesses lower to
// base + immediate-offset loads inside a single clause (no serial addr chains).
// -----------------------------------------------------------------------------
template <int HW>
__global__ __launch_bounds__(256)
void sapa_gemm1x1_wmma_bf16(const float* __restrict__ Wm,
                            const float* __restrict__ bias,
                            const float* __restrict__ img,
                            float* __restrict__ outimg,
                            int M, int Ntot /*B*HW*/)
{
    const int lane = threadIdx.x & 31;
    const int wave = threadIdx.x >> 5;

    const int ntiles = Ntot >> 4;
    const int mtiles = M >> 5;                 // 32-row supertiles
    const int total  = ntiles * mtiles;
    const int t = blockIdx.x * 8 + wave;
    if (t >= total) return;                    // wave-uniform: EXEC all-1 for WMMA
    const int mt = t / ntiles;
    const int nt = t - mt * ntiles;

    const int col  = lane & 15;
    const int half = lane >> 4;                // 0: lanes 0-15, 1: lanes 16-31

    // B-fragment source: column = one pixel, channels strided by HW (constexpr)
    const int gp = nt * 16 + col;              // global pixel over B*HW
    const int b  = gp / HW;                    // HW is a power of two -> shift
    const int p  = gp & (HW - 1);
    const float* __restrict__ xcol = img + (size_t)b * CIN * HW + p;

    // A-fragment sources: two 16-row slabs of W
    const int m0 = mt * 32 + col;
    const float* __restrict__ wrow0 = Wm + (size_t)m0 * CIN;
    const float* __restrict__ wrow1 = wrow0 + (size_t)16 * CIN;

    const int abase = half * 8;                // A K-offset per lane-half
    const int bbase = half * 8;                // B K-offset per lane-half

    v8f acc0 = {}, acc1 = {};
#pragma unroll
    for (int kk = 0; kk < CIN; kk += 32) {
        v16bf a0, a1, bm;
        const float* __restrict__ xk = xcol + (size_t)(kk + bbase) * HW;
#pragma unroll
        for (int e = 0; e < 16; ++e) {
            bm[e] = (bf16_t)xk[(size_t)(e + (e & 8)) * HW]; // const-offset clause
        }
#pragma unroll
        for (int j = 0; j < 8; ++j) {
            a0[j]     = (bf16_t)wrow0[kk + abase + j];
            a0[j + 8] = (bf16_t)wrow0[kk + 16 + abase + j];
            a1[j]     = (bf16_t)wrow1[kk + abase + j];
            a1[j + 8] = (bf16_t)wrow1[kk + 16 + abase + j];
        }
        acc0 = __builtin_amdgcn_wmma_f32_16x16x32_bf16(false, a0, false, bm,
                                                       (short)0, acc0, false, false);
        acc1 = __builtin_amdgcn_wmma_f32_16x16x32_bf16(false, a1, false, bm,
                                                       (short)0, acc1, false, false);
    }

    // C/D layout: VGPR r -> row (+ r + 8*half), col = lane&15
    float* __restrict__ obase = outimg + (size_t)b * M * HW + p;
#pragma unroll
    for (int r = 0; r < 8; ++r) {
        const int mr0 = mt * 32 + r + half * 8;
        const int mr1 = mr0 + 16;
        obase[(size_t)mr0 * HW] = acc0[r] + bias[mr0];
        obase[(size_t)mr1 * HW] = acc1[r] + bias[mr1];
    }
}

// -----------------------------------------------------------------------------
// Deformable attention: one wave per (b, g, hu, wu).
// Phase 1: lanes = embedding channel (E=32). attn[p] = wave-sum(q_e * bilerp(k_e)).
// Phase 2: softmax over the 9 points (lane-uniform registers).
// Phase 3: lanes = value channel (Cg=64, 2 per lane), weighted bilinear gather of x.
// -----------------------------------------------------------------------------
__device__ __forceinline__ float sapa_samp(const float* __restrict__ fp, int y, int x)
{
    if (y < 0 || y >= H_LO || x < 0 || x >= W_LO) return 0.0f;
    return fp[y * W_LO + x];
}

__global__ __launch_bounds__(256)
void sapa_deform_attn(const float* __restrict__ q,
                      const float* __restrict__ k,
                      const float* __restrict__ off,
                      const float* __restrict__ x,
                      float* __restrict__ out, int B)
{
    const int lane = threadIdx.x & 31;
    const int wave = threadIdx.x >> 5;
    const int pos  = blockIdx.x * 8 + wave;
    const int total = B * NGROUPS * HW_UP;
    if (pos >= total) return;

    const int wu = pos & (W_UP - 1);
    int tmp = pos >> 7;                 // / W_UP
    const int hu = tmp & (H_UP - 1);
    tmp >>= 7;                          // / H_UP
    const int g = tmp & (NGROUPS - 1);
    const int b = tmp >> 2;

    const int hl = hu >> 1, wl = wu >> 1;
    const int ij = ((hu & 1) << 1) | (wu & 1);      // pixel-shuffle sub-position

    // q/k channel handled by this lane (embedding dim)
    const float* __restrict__ kg = k + ((size_t)b * (NGROUPS * EDIM) + g * EDIM + lane) * HW_LO;
    const float qe = q[((size_t)b * (NGROUPS * EDIM) + g * EDIM + lane) * HW_UP
                       + (size_t)hu * W_UP + wu];

    const float base_y = (hu + 0.5f) * 0.5f - 0.5f;
    const float base_x = (wu + 0.5f) * 0.5f - 0.5f;

    float attn[NPOINT], pys[NPOINT], pxs[NPOINT];
#pragma unroll
    for (int pnt = 0; pnt < NPOINT; ++pnt) {
        // pixel-shuffled offset channels: ((g*P+pnt)*2 + axis)*4 + ij
        const size_t ob = ((size_t)b * OFFCH + (size_t)(((g * NPOINT + pnt) * 2) * 4 + ij)) * HW_LO
                          + (size_t)hl * W_LO + wl;
        const float py = off[ob] + base_y;
        const float px = off[ob + 4 * HW_LO] + base_x;
        pys[pnt] = py; pxs[pnt] = px;

        const float y0f = floorf(py), x0f = floorf(px);
        const float ty = py - y0f, tx = px - x0f;
        const int y0 = (int)y0f, x0 = (int)x0f;
        const float w00 = (1.0f - ty) * (1.0f - tx);
        const float w01 = (1.0f - ty) * tx;
        const float w10 = ty * (1.0f - tx);
        const float w11 = ty * tx;

        const float ks = w00 * sapa_samp(kg, y0,     x0)
                       + w01 * sapa_samp(kg, y0,     x0 + 1)
                       + w10 * sapa_samp(kg, y0 + 1, x0)
                       + w11 * sapa_samp(kg, y0 + 1, x0 + 1);

        float prod = qe * ks;
#pragma unroll
        for (int o = 16; o > 0; o >>= 1) prod += __shfl_xor(prod, o, 32);
        attn[pnt] = prod;               // identical in all lanes
    }

    // softmax over the 9 points (lane-uniform)
    float mx = attn[0];
#pragma unroll
    for (int pnt = 1; pnt < NPOINT; ++pnt) mx = fmaxf(mx, attn[pnt]);
    float ssum = 0.0f;
#pragma unroll
    for (int pnt = 0; pnt < NPOINT; ++pnt) { attn[pnt] = __expf(attn[pnt] - mx); ssum += attn[pnt]; }
    const float inv = 1.0f / ssum;

    // value gather: Cg = 64 channels, lane handles c = lane and c = lane + 32
    const int Cg = CIN / NGROUPS;
    const float* __restrict__ xg0 = x + ((size_t)b * CIN + g * Cg + lane) * HW_LO;
    const float* __restrict__ xg1 = xg0 + (size_t)32 * HW_LO;

    float acc0 = 0.0f, acc1 = 0.0f;
#pragma unroll
    for (int pnt = 0; pnt < NPOINT; ++pnt) {
        const float py = pys[pnt], px = pxs[pnt];
        const float y0f = floorf(py), x0f = floorf(px);
        const float ty = py - y0f, tx = px - x0f;
        const int y0 = (int)y0f, x0 = (int)x0f;
        const float wsc = attn[pnt] * inv;
        const float w00 = wsc * (1.0f - ty) * (1.0f - tx);
        const float w01 = wsc * (1.0f - ty) * tx;
        const float w10 = wsc * ty * (1.0f - tx);
        const float w11 = wsc * ty * tx;

        acc0 += w00 * sapa_samp(xg0, y0,     x0)
              + w01 * sapa_samp(xg0, y0,     x0 + 1)
              + w10 * sapa_samp(xg0, y0 + 1, x0)
              + w11 * sapa_samp(xg0, y0 + 1, x0 + 1);
        acc1 += w00 * sapa_samp(xg1, y0,     x0)
              + w01 * sapa_samp(xg1, y0,     x0 + 1)
              + w10 * sapa_samp(xg1, y0 + 1, x0)
              + w11 * sapa_samp(xg1, y0 + 1, x0 + 1);
    }

    const size_t obase = ((size_t)b * CIN + g * Cg + lane) * HW_UP + (size_t)hu * W_UP + wu;
    out[obase] = acc0;
    out[obase + (size_t)32 * HW_UP] = acc1;
}

// -----------------------------------------------------------------------------
// Launch
// -----------------------------------------------------------------------------
extern "C" void kernel_launch(void* const* d_in, const int* in_sizes, int n_in,
                              void* d_out, int out_size, void* d_ws, size_t ws_size,
                              hipStream_t stream)
{
    const float* y    = (const float*)d_in[0];
    const float* x    = (const float*)d_in[1];
    const float* Wq   = (const float*)d_in[2];
    const float* bq   = (const float*)d_in[3];
    const float* Wk   = (const float*)d_in[4];
    const float* bk   = (const float*)d_in[5];
    const float* Woff = (const float*)d_in[6];
    const float* boff = (const float*)d_in[7];
    float* out = (float*)d_out;

    const int B = in_sizes[0] / (CIN * HW_UP);   // = 2

    // workspace: q [B,128,HW_UP] | k [B,128,HW_LO] | off [B,288,HW_LO]  (~30.4 MB)
    float* q_ws   = (float*)d_ws;
    float* k_ws   = q_ws + (size_t)B * 128 * HW_UP;
    float* off_ws = k_ws + (size_t)B * 128 * HW_LO;

    { // q = Wq * y + bq
        const int M = 128, N = B * HW_UP;
        const int tiles = (M / 32) * (N / 16);
        sapa_gemm1x1_wmma_bf16<HW_UP><<<(tiles + 7) / 8, 256, 0, stream>>>(Wq, bq, y, q_ws, M, N);
    }
    { // k = Wk * x + bk
        const int M = 128, N = B * HW_LO;
        const int tiles = (M / 32) * (N / 16);
        sapa_gemm1x1_wmma_bf16<HW_LO><<<(tiles + 7) / 8, 256, 0, stream>>>(Wk, bk, x, k_ws, M, N);
    }
    { // off = Woff * x + boff
        const int M = OFFCH, N = B * HW_LO;
        const int tiles = (M / 32) * (N / 16);
        sapa_gemm1x1_wmma_bf16<HW_LO><<<(tiles + 7) / 8, 256, 0, stream>>>(Woff, boff, x, off_ws, M, N);
    }
    { // fused deformable attention
        const int total = B * NGROUPS * HW_UP;
        sapa_deform_attn<<<(total + 7) / 8, 256, 0, stream>>>(q_ws, k_ws, off_ws, x, out, B);
    }
}
